// _ORExplainerCore_20856361189435
// MI455X (gfx1250) — matrix operand order repaired and checked by reading, not compiled, not run on hardware
//
#include <hip/hip_runtime.h>
#include <hip/hip_bf16.h>
#include <math.h>

typedef __attribute__((ext_vector_type(16))) _Float16 v16h;
typedef __attribute__((ext_vector_type(8)))  float    v8f;
typedef __attribute__((ext_vector_type(4)))  float    v4f;

#define EMBED_DIM 320
#define HIDDEN    64
#define NCOLS     128              // [P | Q] per node
#define KSTEP     32
#define KSTEPS    (EMBED_DIM / KSTEP)   // 10
#define KS_HALF   (KSTEPS / 2)          // 5 (two LDS phases, 40KB each)
#define NT        8                     // 128 / 16 N-tiles
#define NNODES    50000

// ------------------------------------------------------------------
// Kernel 0: c_vec[j] = b1[j] + center @ W1[640:960, j]   (64 threads)
// ------------------------------------------------------------------
__global__ __launch_bounds__(64) void cvec_kernel(
    const float* __restrict__ embed, const int* __restrict__ nodeId,
    const float* __restrict__ W1, const float* __restrict__ b1,
    float* __restrict__ cvec)
{
    const int j   = threadIdx.x;                 // 0..63
    const int nid = nodeId[0];
    const float* c = embed + (size_t)nid * EMBED_DIM;
    float acc = b1[j];
    for (int k = 0; k < EMBED_DIM; ++k)
        acc = fmaf(c[k], W1[(size_t)(2 * EMBED_DIM + k) * HIDDEN + j], acc);
    cvec[j] = acc;
}

// ------------------------------------------------------------------
// Kernel 1: R[n][0:64]  = embed[n] @ W1[0:320,:]      (P)
//           R[n][64:128]= embed[n] @ W1[320:640,:]    (Q)
// One wave per 16-node strip; f16 WMMA, f32 accumulate.
// 625 blocks x 160 threads (5 waves) -> 3125 strips exactly.
// ------------------------------------------------------------------
__global__ __launch_bounds__(160) void gemm_pq_kernel(
    const float* __restrict__ embed, const float* __restrict__ W1,
    float* __restrict__ R)
{
    // Fragment-ready B tiles (f16 pairs packed in dwords): 5*8*32*8 dwords = 40KB
    __shared__ __align__(16) unsigned int ldsB[KS_HALF][NT][32][8];

    const int tid   = threadIdx.x;
    const int lane  = tid & 31;
    const int wave  = tid >> 5;                   // 0..4
    const int strip = blockIdx.x * 5 + wave;      // 0..3124
    const int m     = lane & 15;                  // A-row / C-col within tile
    const int kh    = lane >> 4;                  // K-half selector

    const float* arow = embed + (size_t)(strip * 16 + m) * EMBED_DIM;

    v8f acc[NT];
    const v8f vzero = {0.f, 0.f, 0.f, 0.f, 0.f, 0.f, 0.f, 0.f};
#pragma unroll
    for (int t = 0; t < NT; ++t) acc[t] = vzero;

    for (int phase = 0; phase < 2; ++phase) {
        __syncthreads();   // protect previous phase's LDS reads
        // ---- stage B fragments for K-steps [phase*5, phase*5+5) ----
        for (int idx = tid; idx < KS_HALF * NT * 32 * 8; idx += 160) {
            const int d   = idx & 7;               // dword within fragment
            const int ln  = (idx >> 3) & 31;       // owning lane
            const int t   = (idx >> 8) & 7;        // N-tile
            const int ksl = idx >> 11;             // local K-step 0..4
            const int lkh = ln >> 4;
            const int n   = (t << 4) | (ln & 15);  // global N 0..127
            const int K0  = (phase * KS_HALF + ksl) * KSTEP + lkh * 8 + (d & 3) * 2 + (d >> 2) * 16;
            // Wcat[K][n]: n<64 -> W1 rows 0..319 ; n>=64 -> W1 rows 320..639
            const float* src = (n < HIDDEN)
                ? (W1 + (size_t)K0 * HIDDEN + n)
                : (W1 + (size_t)(EMBED_DIM + K0) * HIDDEN + (n - HIDDEN));
            const _Float16 h0 = (_Float16)src[0];
            const _Float16 h1 = (_Float16)src[HIDDEN];
            const unsigned int u =
                  (unsigned int)__builtin_bit_cast(unsigned short, h0)
                | ((unsigned int)__builtin_bit_cast(unsigned short, h1) << 16);
            ldsB[ksl][t][ln][d] = u;
        }
        __syncthreads();

        // ---- 5 K-steps of 8 WMMA tiles each ----
#pragma unroll
        for (int ksl = 0; ksl < KS_HALF; ++ksl) {
            const int kb = (phase * KS_HALF + ksl) * KSTEP;
            // A fragment (16x32 f16): lane<16 holds K kb+0..7 & kb+16..23,
            // lane>=16 holds K kb+8..15 & kb+24..31 (documented layout).
            const float* p1 = arow + kb + kh * 8;
            const float* p2 = p1 + 16;
            const v4f a0 = *(const v4f*)(p1);
            const v4f a1 = *(const v4f*)(p1 + 4);
            const v4f b0 = *(const v4f*)(p2);
            const v4f b1v = *(const v4f*)(p2 + 4);
            v16h A;
#pragma unroll
            for (int r = 0; r < 4; ++r) {
                A[r]      = (_Float16)a0[r];
                A[4 + r]  = (_Float16)a1[r];
                A[8 + r]  = (_Float16)b0[r];
                A[12 + r] = (_Float16)b1v[r];
            }
#pragma unroll
            for (int t = 0; t < NT; ++t) {
                const v16h B = *reinterpret_cast<const v16h*>(&ldsB[ksl][t][lane][0]);
                acc[t] = __builtin_amdgcn_wmma_f32_16x16x32_f16(
                    false, A, false, B, (short)0, acc[t], false, false);
            }
        }
    }

    // C layout: lane<16 -> M = vgpr_i, lane>=16 -> M = 8 + vgpr_i; N = lane%16
    float* outp = R + (size_t)(strip * 16 + kh * 8) * NCOLS + m;
#pragma unroll
    for (int t = 0; t < NT; ++t)
#pragma unroll
        for (int i = 0; i < 8; ++i)
            outp[(size_t)i * NCOLS + t * 16] = acc[t][i];
}

// ------------------------------------------------------------------
// Kernel 2: per-edge gate.  One thread per edge; P/Q table lives in L2.
// ------------------------------------------------------------------
__global__ __launch_bounds__(256) void edge_kernel(
    const int* __restrict__ edge_index, const float* __restrict__ eps,
    const float* __restrict__ R, const float* __restrict__ cvec,
    const float* __restrict__ W2, const float* __restrict__ b2,
    const float* __restrict__ tmpPtr, float* __restrict__ out, int E)
{
    __shared__ float s_c[HIDDEN];
    __shared__ float s_w[HIDDEN];
    const int tid = threadIdx.x;
    if (tid < HIDDEN) { s_c[tid] = cvec[tid]; s_w[tid] = W2[tid]; }
    __syncthreads();

    const int e = blockIdx.x * blockDim.x + tid;
    if (e >= E) return;

    const int col = edge_index[e];
    const int row = edge_index[E + e];
    const float* pr = R + (size_t)col * NCOLS;            // P[col]
    const float* qr = R + (size_t)row * NCOLS + HIDDEN;   // Q[row]

    float acc = 0.f;
#pragma unroll
    for (int j4 = 0; j4 < HIDDEN / 4; ++j4) {
        const v4f p = *(const v4f*)(pr + j4 * 4);
        const v4f q = *(const v4f*)(qr + j4 * 4);
#pragma unroll
        for (int r = 0; r < 4; ++r) {
            const int j = j4 * 4 + r;
            float v = p[r] + q[r] + s_c[j];
            v = v > 0.f ? v : 0.f;
            acc = fmaf(v, s_w[j], acc);
        }
    }
    const float w = acc + b2[0];

    const float bias = 1.0e-4f;                     // SAMPLE_BIAS + 1e-4
    const float ep   = eps[e] * (1.f - 2.f * bias) + bias;
    const float gate = (logf(ep) - log1pf(-ep) + w) / tmpPtr[0];
    out[e] = 1.f / (1.f + expf(-gate));
}

// ------------------------------------------------------------------
extern "C" void kernel_launch(void* const* d_in, const int* in_sizes, int n_in,
                              void* d_out, int out_size, void* d_ws, size_t ws_size,
                              hipStream_t stream)
{
    // setup_inputs order: x, embed, edge_index, node_id, tmp, eps, W1, b1, W2, b2
    const float* embed = (const float*)d_in[1];
    const int*   eidx  = (const int*)d_in[2];
    const int*   nid   = (const int*)d_in[3];
    const float* tmp   = (const float*)d_in[4];
    const float* eps   = (const float*)d_in[5];
    const float* W1    = (const float*)d_in[6];
    const float* b1    = (const float*)d_in[7];
    const float* W2    = (const float*)d_in[8];
    const float* b2    = (const float*)d_in[9];
    float* out = (float*)d_out;
    const int E = in_sizes[5];                      // eps length = #edges

    // workspace: R = 50000 x 128 f32 (25.6 MB), then cvec (64 f32)
    float* R    = (float*)d_ws;
    float* cvec = R + (size_t)NNODES * NCOLS;

    cvec_kernel<<<1, 64, 0, stream>>>(embed, nid, W1, b1, cvec);

    const int nStrips = NNODES / 16;                // 3125, divides as 625*5
    gemm_pq_kernel<<<nStrips / 5, 160, 0, stream>>>(embed, W1, R);

    edge_kernel<<<(E + 255) / 256, 256, 0, stream>>>(
        eidx, eps, R, cvec, W2, b2, tmp, out, E);
}
